// FCEncoder_31499290149430
// MI455X (gfx1250) — matrix operand order repaired and checked by reading, not compiled, not run on hardware
//
#include <hip/hip_runtime.h>
#include <hip/hip_bf16.h>
#include <math.h>

// ---------- problem sizes ----------
#define BQ   16
#define TT   512
#define CC   1024
#define FF   2048
#define OUTC 128

// ---------- GEMM tiling ----------
#define BM   128
#define BN   64
#define BKK  32
#define LSTR 40   // LDS row stride in bf16 elements (padded: 16 distinct banks across rows)

typedef __attribute__((ext_vector_type(16))) __bf16 v16bf;
typedef __attribute__((ext_vector_type(8)))  float  v8f;

#define WMMA_BF16(a, b, c) \
  __builtin_amdgcn_wmma_f32_16x16x32_bf16(false, (a), false, (b), (short)0, (c), false, false)

__device__ __forceinline__ unsigned short f2bf(float x) {
  union { float f; unsigned u; } v; v.f = x;
  unsigned r = v.u + 0x7fffu + ((v.u >> 16) & 1u);   // RNE
  return (unsigned short)(r >> 16);
}

union FragU { v16bf v; unsigned u[8]; };

// A fragment: 16x32 bf16 tile in LDS (row stride LSTR). ISA layout:
// lanes 0-15: row M=lane, K = {0..7, 16..23}; lanes 16-31: K = {8..15, 24..31}.
__device__ __forceinline__ v16bf load_a_frag(const unsigned short* s, int lane) {
  const int r    = lane & 15;
  const int koff = (lane >> 4) << 3;          // 0 or 8
  const unsigned short* row = s + r * LSTR + koff;
  FragU f;
#pragma unroll
  for (int j = 0; j < 4; ++j) f.u[j]     = *(const unsigned*)(row + 2 * j);
#pragma unroll
  for (int j = 0; j < 4; ++j) f.u[4 + j] = *(const unsigned*)(row + 16 + 2 * j);
  return f.v;
}

// B fragment from Bt (N x K row-major) tile in LDS: lane holds column n=lane&15,
// K = {0..7, 16..23} (lanes 0-15) or {8..15, 24..31} (lanes 16-31), packed pairs.
__device__ __forceinline__ v16bf load_b_frag(const unsigned short* s, int lane) {
  const int n  = lane & 15;
  const int kb = (lane >> 4) << 3;            // 0 or 8
  const unsigned short* row = s + n * LSTR + kb;
  FragU f;
#pragma unroll
  for (int j = 0; j < 4; ++j) f.u[j]     = *(const unsigned*)(row + 2 * j);
#pragma unroll
  for (int j = 0; j < 4; ++j) f.u[4 + j] = *(const unsigned*)(row + 16 + 2 * j);
  return f.v;
}

// 128x64 block tile, 8 waves, each wave owns 32x32 (4 accumulators).
// Double-buffered LDS: one barrier per K-step; next tile staged in registers
// while WMMAs consume the current buffer.
__device__ __forceinline__ void gemm128x64(
    const unsigned short* __restrict__ A, int lda,
    const unsigned short* __restrict__ Bt, int ldb,
    int K, int m0, int n0, v8f c[2][2])
{
  __shared__ unsigned short As[2][BM * LSTR];
  __shared__ unsigned short Bs[2][BN * LSTR];
  const int tid  = threadIdx.x;
  const int lane = tid & 31, wave = tid >> 5;
  const int rb = (wave & 3) << 5;             // wave row base: 0/32/64/96
  const int cb = (wave >> 2) << 5;            // wave col base: 0/32
  const int lr = tid >> 2;                    // loader row 0..63
  const int lc = (tid & 3) << 3;              // loader col (8 bf16 = 16B)

  uint4 ga0 = *(const uint4*)&A[(long)(m0 + lr)      * lda + lc];
  uint4 ga1 = *(const uint4*)&A[(long)(m0 + lr + 64) * lda + lc];
  uint4 gb  = *(const uint4*)&Bt[(long)(n0 + lr)     * ldb + lc];

  int buf = 0;
  for (int k0 = 0; k0 < K; k0 += BKK) {
    *(uint4*)&As[buf][lr * LSTR + lc]        = ga0;
    *(uint4*)&As[buf][(lr + 64) * LSTR + lc] = ga1;
    *(uint4*)&Bs[buf][lr * LSTR + lc]        = gb;
    __syncthreads();
    const int kn = k0 + BKK;
    if (kn < K) {
      ga0 = *(const uint4*)&A[(long)(m0 + lr)      * lda + kn + lc];
      ga1 = *(const uint4*)&A[(long)(m0 + lr + 64) * lda + kn + lc];
      gb  = *(const uint4*)&Bt[(long)(n0 + lr)     * ldb + kn + lc];
    }
    v16bf a0 = load_a_frag(&As[buf][rb * LSTR],        lane);
    v16bf a1 = load_a_frag(&As[buf][(rb + 16) * LSTR], lane);
    v16bf b0 = load_b_frag(&Bs[buf][cb * LSTR],        lane);
    v16bf b1 = load_b_frag(&Bs[buf][(cb + 16) * LSTR], lane);
    c[0][0] = WMMA_BF16(a0, b0, c[0][0]);
    c[0][1] = WMMA_BF16(a0, b1, c[0][1]);
    c[1][0] = WMMA_BF16(a1, b0, c[1][0]);
    c[1][1] = WMMA_BF16(a1, b1, c[1][1]);
    buf ^= 1;
  }
}

// C/D layout: col n = lane&15, row m = vgpr + 8*(lane>=16); wave tile 32x32.
#define EPI_LOOP(...)                                                              \
  { const int lane = threadIdx.x & 31, wave = threadIdx.x >> 5;                    \
    const int rb = (wave & 3) << 5; const int cb = (wave >> 2) << 5;               \
    const int nc = lane & 15; const int mh = (lane >> 4) << 3;                     \
    _Pragma("unroll")                                                              \
    for (int im = 0; im < 2; ++im) {                                               \
      _Pragma("unroll")                                                            \
      for (int v = 0; v < 8; ++v) {                                                \
        const int m = m0 + rb + im * 16 + mh + v;                                  \
        { const int n = n0 + cb + nc;      const float accv = c[im][0][v]; __VA_ARGS__ } \
        { const int n = n0 + cb + 16 + nc; const float accv = c[im][1][v]; __VA_ARGS__ } \
      } } }

#define ACC_INIT v8f c[2][2]; \
  { _Pragma("unroll") for (int i = 0; i < 2; ++i) \
    _Pragma("unroll") for (int j = 0; j < 2; ++j) \
    _Pragma("unroll") for (int v = 0; v < 8; ++v) c[i][j][v] = 0.f; }

// ---------------- prep kernels ----------------

// xm[e][b][t][c] = bf16( t<len[b] ? inputs[b][t][e?c:1024+c] : 0 )
__global__ void k_prep_x(const float* __restrict__ inputs,
                         const int* __restrict__ lens,
                         unsigned short* __restrict__ xm)
{
  const long i = (long)blockIdx.x * 256 + threadIdx.x;   // 2^24 total
  const int c = (int)(i & 1023);
  const int t = (int)((i >> 10) & 511);
  const int b = (int)((i >> 19) & 15);
  const int e = (int)(i >> 23);
  float v = 0.f;
  if (t < lens[b]) v = inputs[((long)b * TT + t) * FF + (e == 0 ? CC : 0) + c];
  xm[i] = f2bf(v);
}

// conv weights (O,I,3) -> Bt layout wt[which][o][dk*1024 + i]
__global__ void k_prep_wt(const float* __restrict__ wq,
                          const float* __restrict__ wk,
                          const float* __restrict__ wv,
                          unsigned short* __restrict__ wt)
{
  const long i = (long)blockIdx.x * 256 + threadIdx.x;   // 3 * 1024 * 3072
  const int ci = (int)(i & 1023);
  const long j = i >> 10;                                 // which*3072 + o*3 + dk
  const int dk = (int)(j % 3);
  const int o  = (int)((j / 3) & 1023);
  const int which = (int)((j / 3) >> 10);
  const float* src = (which == 0) ? wq : (which == 1 ? wk : wv);
  wt[i] = f2bf(src[(long)o * 3072 + ci * 3 + dk]);
}

__global__ void k_cvt(const float* __restrict__ s, unsigned short* __restrict__ d, int n)
{
  const int i = blockIdx.x * 256 + threadIdx.x;
  if (i < n) d[i] = f2bf(s[i]);
}

// ---------------- conv-as-GEMM: Q/K/V ----------------
__global__ __launch_bounds__(256)
void k_conv_qkv(const unsigned short* __restrict__ xm,
                const unsigned short* __restrict__ wt,
                const float* __restrict__ bq, const float* __restrict__ bk,
                const float* __restrict__ bv,
                unsigned short* __restrict__ Q,
                unsigned short* __restrict__ Kb,
                unsigned short* __restrict__ Vt)
{
  const int z = blockIdx.z;                 // ((e*3+which)*16 + b)
  const int b = z & 15;
  const int ew = z >> 4;
  const int which = ew % 3;
  const int e = ew / 3;
  const unsigned short* X  = xm + (long)(e * BQ + b) * (TT * CC);
  const unsigned short* Bt = wt + (long)which * (CC * 3 * CC);
  const int m0 = blockIdx.y * BM, n0 = blockIdx.x * BN;
  ACC_INIT
  {
    __shared__ unsigned short As[2][BM * LSTR];
    __shared__ unsigned short Bs[2][BN * LSTR];
    const int tid = threadIdx.x;
    const int lane = tid & 31, wave = tid >> 5;
    const int rb = (wave & 3) << 5, cb = (wave >> 2) << 5;
    const int lr = tid >> 2, lc = (tid & 3) << 3;

    // conv A-loader: k = dk*1024 + i  ->  A(t,k) = X[t+dk-1, i], zero-padded in time
    uint4 ga0, ga1, gb;
    {
      const int dk = 0, i0 = lc;
      const int t0 = m0 + lr + dk - 1, t1 = t0 + 64;
      ga0 = {0u,0u,0u,0u}; ga1 = {0u,0u,0u,0u};
      if (t0 >= 0 && t0 < TT) ga0 = *(const uint4*)&X[(long)t0 * CC + i0];
      if (t1 >= 0 && t1 < TT) ga1 = *(const uint4*)&X[(long)t1 * CC + i0];
      gb = *(const uint4*)&Bt[(long)(n0 + lr) * (3 * CC) + lc];
    }
    int buf = 0;
    for (int k0 = 0; k0 < 3 * CC; k0 += BKK) {
      *(uint4*)&As[buf][lr * LSTR + lc]        = ga0;
      *(uint4*)&As[buf][(lr + 64) * LSTR + lc] = ga1;
      *(uint4*)&Bs[buf][lr * LSTR + lc]        = gb;
      __syncthreads();
      const int kn = k0 + BKK;
      if (kn < 3 * CC) {
        const int dk = kn >> 10;               // BKK never straddles a tap (1024 % 32 == 0)
        const int i0 = (kn & (CC - 1)) + lc;
        const int t0 = m0 + lr + dk - 1, t1 = t0 + 64;
        ga0 = {0u,0u,0u,0u}; ga1 = {0u,0u,0u,0u};
        if (t0 >= 0 && t0 < TT) ga0 = *(const uint4*)&X[(long)t0 * CC + i0];
        if (t1 >= 0 && t1 < TT) ga1 = *(const uint4*)&X[(long)t1 * CC + i0];
        gb = *(const uint4*)&Bt[(long)(n0 + lr) * (3 * CC) + kn + lc];
      }
      v16bf a0 = load_a_frag(&As[buf][rb * LSTR],        lane);
      v16bf a1 = load_a_frag(&As[buf][(rb + 16) * LSTR], lane);
      v16bf b0 = load_b_frag(&Bs[buf][cb * LSTR],        lane);
      v16bf b1 = load_b_frag(&Bs[buf][(cb + 16) * LSTR], lane);
      c[0][0] = WMMA_BF16(a0, b0, c[0][0]);
      c[0][1] = WMMA_BF16(a0, b1, c[0][1]);
      c[1][0] = WMMA_BF16(a1, b0, c[1][0]);
      c[1][1] = WMMA_BF16(a1, b1, c[1][1]);
      buf ^= 1;
    }
  }
  const float* bias = (which == 0) ? bq : (which == 1 ? bk : bv);
  const long ob = (long)(e * BQ + b) * (TT * CC);
  EPI_LOOP({
    const float val = accv + bias[n];
    if (which == 2)      Vt[ob + (long)n * TT + m] = f2bf(val);   // V stored transposed (c,s)
    else if (which == 0) Q [ob + (long)m * CC + n] = f2bf(val);
    else                 Kb[ob + (long)m * CC + n] = f2bf(val);
  })
}

// ---------------- scores = Q K^T / sqrt(len), masked ----------------
__global__ __launch_bounds__(256)
void k_scores(const unsigned short* __restrict__ Q,
              const unsigned short* __restrict__ Kb,
              const int* __restrict__ lens,
              float* __restrict__ sc)
{
  const int z = blockIdx.z; const int b = z & 15;
  const unsigned short* A  = Q  + (long)z * (TT * CC);
  const unsigned short* Bt = Kb + (long)z * (TT * CC);
  const int m0 = blockIdx.y * BM, n0 = blockIdx.x * BN;
  ACC_INIT
  gemm128x64(A, CC, Bt, CC, CC, m0, n0, c);
  const int len = lens[b];
  const float rs = rsqrtf((float)len);
  float* out = sc + (long)z * (TT * TT);
  EPI_LOOP({
    out[(long)m * TT + n] = (n < len) ? accv * rs : -INFINITY;
  })
}

// ---------------- row softmax -> bf16 attn ----------------
__global__ __launch_bounds__(256)
void k_softmax(const float* __restrict__ sc, unsigned short* __restrict__ attn)
{
  const long row = blockIdx.x;               // 2*16*512 rows of length 512
  const float* src = sc + row * TT;
  __shared__ float red[256];
  const int tid = threadIdx.x;
  const float a = src[tid], b = src[tid + 256];
  red[tid] = fmaxf(a, b);
  __syncthreads();
  for (int s = 128; s > 0; s >>= 1) { if (tid < s) red[tid] = fmaxf(red[tid], red[tid + s]); __syncthreads(); }
  const float m = red[0];
  __syncthreads();
  const float ea = __expf(a - m), eb = __expf(b - m);   // exp(-inf)=0 handles mask
  red[tid] = ea + eb;
  __syncthreads();
  for (int s = 128; s > 0; s >>= 1) { if (tid < s) red[tid] += red[tid + s]; __syncthreads(); }
  const float inv = 1.0f / red[0];
  unsigned short* dst = attn + row * TT;
  dst[tid]       = f2bf(ea * inv);
  dst[tid + 256] = f2bf(eb * inv);
}

// ---------------- out = attn @ V ; a = mask ? gamma*out + x : x ----------------
__global__ __launch_bounds__(256)
void k_attout(const unsigned short* __restrict__ attn,
              const unsigned short* __restrict__ Vt,
              const float* __restrict__ inputs,
              const int* __restrict__ lens,
              const float* __restrict__ gammap,
              unsigned short* __restrict__ aout)
{
  const int z = blockIdx.z; const int e = z >> 4; const int b = z & 15;
  const unsigned short* A  = attn + (long)z * (TT * TT);   // (t,s)
  const unsigned short* Bt = Vt   + (long)z * (TT * CC);   // (c,s) row-major
  const int m0 = blockIdx.y * BM, n0 = blockIdx.x * BN;
  ACC_INIT
  gemm128x64(A, TT, Bt, TT, TT, m0, n0, c);
  const float gamma = gammap[0];
  const int len = lens[b];
  const int choff = (e == 0) ? CC : 0;
  const float* x = inputs + (long)b * TT * FF;
  unsigned short* dst = aout + (long)z * (TT * CC);
  EPI_LOOP({
    const float xv = x[(long)m * FF + choff + n];
    const float val = (m < len) ? gamma * accv + xv : xv;   // xm==x where masked
    dst[(long)m * CC + n] = f2bf(val);
  })
}

// ---------------- head layer 1: h = relu(a W1^T + b1) ----------------
__global__ __launch_bounds__(256)
void k_head1(const unsigned short* __restrict__ a,
             const unsigned short* __restrict__ wfc,
             const float* __restrict__ bf, const float* __restrict__ br,
             unsigned short* __restrict__ h)
{
  const int e = blockIdx.z;
  const unsigned short* A  = a   + (long)e * (BQ * TT * CC);
  const unsigned short* Bt = wfc + (long)e * (CC * CC);
  const int m0 = blockIdx.y * BM, n0 = blockIdx.x * BN;
  ACC_INIT
  gemm128x64(A, CC, Bt, CC, CC, m0, n0, c);
  const float* bias = e ? br : bf;
  unsigned short* dst = h + (long)e * (BQ * TT * CC);
  EPI_LOOP({
    dst[(long)m * CC + n] = f2bf(fmaxf(accv + bias[n], 0.f));
  })
}

// ---------------- head layer 2: y = relu(h W2^T + b2) -> concat f32 out ----------------
__global__ __launch_bounds__(256)
void k_head2(const unsigned short* __restrict__ h,
             const unsigned short* __restrict__ w1,
             const float* __restrict__ bf, const float* __restrict__ br,
             float* __restrict__ out)
{
  const int e = blockIdx.z;
  const unsigned short* A  = h  + (long)e * (BQ * TT * CC);
  const unsigned short* Bt = w1 + (long)e * (OUTC * CC);
  const int m0 = blockIdx.y * BM, n0 = blockIdx.x * BN;
  ACC_INIT
  gemm128x64(A, CC, Bt, CC, CC, m0, n0, c);
  const float* bias = e ? br : bf;
  EPI_LOOP({
    out[(long)m * (2 * OUTC) + e * OUTC + n] = fmaxf(accv + bias[n], 0.f);
  })
}

// ---------------- launcher ----------------
extern "C" void kernel_launch(void* const* d_in, const int* in_sizes, int n_in,
                              void* d_out, int out_size, void* d_ws, size_t ws_size,
                              hipStream_t stream) {
  const float* inputs  = (const float*)d_in[0];
  const float* wq      = (const float*)d_in[1];
  const float* bq      = (const float*)d_in[2];
  const float* wk      = (const float*)d_in[3];
  const float* bk      = (const float*)d_in[4];
  const float* wv      = (const float*)d_in[5];
  const float* bv      = (const float*)d_in[6];
  const float* gamma   = (const float*)d_in[7];
  const float* w_fcf   = (const float*)d_in[8];
  const float* b_fcf   = (const float*)d_in[9];
  const float* w_fc1f  = (const float*)d_in[10];
  const float* b_fc1f  = (const float*)d_in[11];
  const float* w_fcr   = (const float*)d_in[12];
  const float* b_fcr   = (const float*)d_in[13];
  const float* w_fc1r  = (const float*)d_in[14];
  const float* b_fc1r  = (const float*)d_in[15];
  const int*   lens    = (const int*)d_in[16];
  float*       out     = (float*)d_out;

  char* ws = (char*)d_ws;
  unsigned short* xm   = (unsigned short*)(ws + 0);           //  32 MB  xm bf16 (2,B,T,C)
  unsigned short* wt   = (unsigned short*)(ws + 33554432);    //  18 MB  conv W^T bf16
  unsigned short* wfc  = (unsigned short*)(ws + 52428800);    //   4 MB  fc W bf16
  unsigned short* wfc1 = (unsigned short*)(ws + 56623104);    // 0.5 MB  fc1 W bf16
  unsigned short* Qb   = (unsigned short*)(ws + 57147392);    //  32 MB
  unsigned short* Kbuf = (unsigned short*)(ws + 90701824);    //  32 MB
  unsigned short* Vt   = (unsigned short*)(ws + 124256256);   //  32 MB  V transposed (c,s)
  float*          sc   = (float*)(ws + 157810688);            //  64 MB  scores f32
  unsigned short* attn = (unsigned short*)(ws + 224919552);   //  16 MB  attn bf16
  unsigned short* aout = (unsigned short*)(ws + 157810688);   //  reuse scores (dead after softmax)
  unsigned short* hbuf = (unsigned short*)(ws + 157810688 + 33554432);

  k_prep_x <<<65536, 256, 0, stream>>>(inputs, lens, xm);
  k_prep_wt<<<36864, 256, 0, stream>>>(wq, wk, wv, wt);
  k_cvt<<<4096, 256, 0, stream>>>(w_fcf,  wfc,            1048576);
  k_cvt<<<4096, 256, 0, stream>>>(w_fcr,  wfc + 1048576,  1048576);
  k_cvt<<< 512, 256, 0, stream>>>(w_fc1f, wfc1,           131072);
  k_cvt<<< 512, 256, 0, stream>>>(w_fc1r, wfc1 + 131072,  131072);

  k_conv_qkv<<<dim3(16, 4, 96), 256, 0, stream>>>(xm, wt, bq, bk, bv, Qb, Kbuf, Vt);
  k_scores  <<<dim3(8,  4, 32), 256, 0, stream>>>(Qb, Kbuf, lens, sc);
  k_softmax <<<16384, 256, 0, stream>>>(sc, attn);
  k_attout  <<<dim3(16, 4, 32), 256, 0, stream>>>(attn, Vt, inputs, lens, gamma, aout);
  k_head1   <<<dim3(16, 64, 2), 256, 0, stream>>>(aout, wfc, b_fcf, b_fcr, hbuf);
  k_head2   <<<dim3(2,  64, 2), 256, 0, stream>>>(hbuf, wfc1, b_fc1f, b_fc1r, out);
}
